// MotionFormerFullsize_90855738180248
// MI455X (gfx1250) — matrix operand order repaired and checked by reading, not compiled, not run on hardware
//
#include <hip/hip_runtime.h>
#include <hip/hip_bf16.h>

typedef __attribute__((ext_vector_type(16))) __bf16 v16bf;
typedef __attribute__((ext_vector_type(8)))  __bf16 v8bf;
typedef __attribute__((ext_vector_type(8)))  float  v8f;

union BF16x16 { v16bf v; v8bf h[2]; __bf16 e[16]; };

__device__ __forceinline__ float gelu_exact(float x) {
    return 0.5f * x * (1.0f + erff(x * 0.70710678118654752440f));
}

// ---------------------------------------------------------------------------
// CDNA5 async global->LDS copies (tracked by ASYNCcnt). Generic pointers to
// __shared__ truncate to the 32-bit LDS offset (ISA 10.2).
// ---------------------------------------------------------------------------
__device__ __forceinline__ void async_ld_b128(void* lds, const void* gaddr) {
    unsigned off = (unsigned)(unsigned long long)lds;
    asm volatile("global_load_async_to_lds_b128 %0, %1, off"
                 :: "v"(off), "v"(gaddr) : "memory");
}
__device__ __forceinline__ void async_ld_b32(void* lds, const void* gaddr) {
    unsigned off = (unsigned)(unsigned long long)lds;
    asm volatile("global_load_async_to_lds_b32 %0, %1, off"
                 :: "v"(off), "v"(gaddr) : "memory");
}
__device__ __forceinline__ void wait_async0() {
#if __has_builtin(__builtin_amdgcn_s_wait_asynccnt)
    __builtin_amdgcn_s_wait_asynccnt(0);
#else
    asm volatile("s_wait_asynccnt 0x0" ::: "memory");
#endif
}

// ---------------------------------------------------------------------------
// Tensor Data Mover: DMA a 2D tile (tile_dim0=32 elems of 2B, tile_dim1=128
// rows, row stride = strideElems) from global into LDS. D# built per ISA
// CDNA5 ISA. Issued once per workgroup (wave 0), TENSORcnt.
// amdgpu-toolchain (clang-23) builtin arity: 6 args
//   (uint32x4 g0, int32x8 g1, int32x4, int32x4, int32x8, i32 cpol)
// ---------------------------------------------------------------------------
#if __has_builtin(__builtin_amdgcn_tensor_load_to_lds) && __has_builtin(__builtin_amdgcn_s_wait_tensorcnt)
#define HAVE_TDM 1
#endif

#ifdef HAVE_TDM
typedef unsigned int u32x4 __attribute__((ext_vector_type(4)));
typedef int i32x8 __attribute__((ext_vector_type(8)));
typedef int i32x4 __attribute__((ext_vector_type(4)));

__device__ __forceinline__ void tdm_load_tile_128x32(void* lds, const void* gaddr,
                                                     unsigned strideElems)
{
    const unsigned long long ga = (unsigned long long)gaddr;
    // group0: count=1 | lds_addr | global_addr[56:0] | type=2 (bits 127:126)
    u32x4 g0 = { 1u,
                 (unsigned)(unsigned long long)lds,
                 (unsigned)ga,
                 (unsigned)((ga >> 32) & 0x1FFFFFFu) | 0x80000000u };
    // group1: data_size=2B; tensor_dim0=tensor_dim1=2^30 (no OOB clipping);
    // tile_dim0=32, tile_dim1=128, tile_dim2=0; tensor_dim0_stride=strideElems
    i32x8 g1 = { 0x00010000,          // wgmask=0, data_size=1 (2B)
                 0x00000000,          // atomic_barrier=0, tensor_dim0[15:0]=0
                 0x00004000,          // tensor_dim0[31:16]=0x4000, tensor_dim1[15:0]=0
                 0x00204000,          // tensor_dim1[31:16]=0x4000, tile_dim0=32
                 128,                 // tile_dim1=128, tile_dim2=0
                 (int)strideElems,    // tensor_dim0_stride[31:0]
                 0, 0 };
    i32x4 z4 = { 0, 0, 0, 0 };
    i32x8 z8 = { 0, 0, 0, 0, 0, 0, 0, 0 };
    __builtin_amdgcn_tensor_load_to_lds(g0, g1, z4, z4, z8, 0);
}
__device__ __forceinline__ void wait_tensor0() {
    __builtin_amdgcn_s_wait_tensorcnt((short)0);
}
#endif

// ---------------------------------------------------------------------------
// Row LayerNorm over 256 channels (std_ln: biased var, /sqrt(var+eps)).
// ---------------------------------------------------------------------------
__global__ __launch_bounds__(256)
void ln256_kernel(const float* __restrict__ x, const float* __restrict__ w,
                  const float* __restrict__ bias,
                  float* __restrict__ outF, __bf16* __restrict__ outB)
{
    const int row  = blockIdx.x * 8 + (threadIdx.x >> 5);
    const int lane = threadIdx.x & 31;
    const float* xp = x + (size_t)row * 256 + lane * 8;
    float v[8];
    {
        float4 p0 = *(const float4*)xp;
        float4 p1 = *(const float4*)(xp + 4);
        v[0]=p0.x; v[1]=p0.y; v[2]=p0.z; v[3]=p0.w;
        v[4]=p1.x; v[5]=p1.y; v[6]=p1.z; v[7]=p1.w;
    }
    float s = 0.f, ss = 0.f;
    #pragma unroll
    for (int i = 0; i < 8; ++i) { s += v[i]; ss += v[i]*v[i]; }
    #pragma unroll
    for (int off = 16; off > 0; off >>= 1) {
        s  += __shfl_xor(s,  off, 32);
        ss += __shfl_xor(ss, off, 32);
    }
    const float mean = s * (1.0f/256.0f);
    const float var  = ss * (1.0f/256.0f) - mean*mean;
    const float rstd = rsqrtf(var + 1e-5f);
    #pragma unroll
    for (int i = 0; i < 8; ++i) {
        const int c = lane*8 + i;
        const float o = w[c] * (v[i] - mean) * rstd + bias[c];
        if (outF) outF[(size_t)row*256 + c] = o;
        if (outB) outB[(size_t)row*256 + c] = (__bf16)o;
    }
}

// ---------------------------------------------------------------------------
// Per-head custom LayerNorm (ddof=1, /(std+eps)) on k and v halves of kv.
// ---------------------------------------------------------------------------
__global__ __launch_bounds__(256)
void lnhead_kernel(const float* __restrict__ kv,
                   const float* __restrict__ klw, const float* __restrict__ klb,
                   const float* __restrict__ vlw, const float* __restrict__ vlb,
                   __bf16* __restrict__ k_bf, __bf16* __restrict__ v_bf)
{
    const int gid = blockIdx.x * blockDim.x + threadIdx.x;   // 65536*16
    const int row = gid >> 4;
    const int p   = gid & 15;
    const int isV = p >> 3;
    const int hh  = p & 7;
    const float* src = kv + (size_t)row * 512 + isV * 256 + hh * 32;
    float vals[32]; float s = 0.f;
    #pragma unroll
    for (int i = 0; i < 32; ++i) { vals[i] = src[i]; s += vals[i]; }
    const float mean = s * (1.0f/32.0f);
    float ss = 0.f;
    #pragma unroll
    for (int i = 0; i < 32; ++i) { float d = vals[i] - mean; ss += d*d; }
    const float stdv = sqrtf(ss * (1.0f/31.0f));     // unbiased
    const float inv  = 1.0f / (stdv + 1e-5f);
    const float* wv = (isV ? vlw : klw) + hh * 32;
    const float* bv = (isV ? vlb : klb) + hh * 32;
    __bf16* dst = (isV ? v_bf : k_bf) + (size_t)row * 256 + hh * 32;
    #pragma unroll
    for (int i = 0; i < 32; ++i)
        dst[i] = (__bf16)(wv[i] * (vals[i] - mean) * inv + bv[i]);
}

// ---------------------------------------------------------------------------
// ce_flat = corw @ cor_w^T (K=2), plus per-head custom LN over md=16.
// ---------------------------------------------------------------------------
__global__ __launch_bounds__(256)
void ce_kernel(const float* __restrict__ cor, const float* __restrict__ cor_w,
               const float* __restrict__ clw, const float* __restrict__ clb,
               float* __restrict__ ce_f32, __bf16* __restrict__ ce_bf)
{
    const int gid = blockIdx.x * blockDim.x + threadIdx.x;   // 65536*8
    const int row = gid >> 3;
    const int hh  = gid & 7;
    const float cx = cor[(size_t)row*2], cy = cor[(size_t)row*2 + 1];
    float vals[16]; float s = 0.f;
    #pragma unroll
    for (int m = 0; m < 16; ++m) {
        const int mm = hh*16 + m;
        const float val = cx * cor_w[mm*2] + cy * cor_w[mm*2 + 1];
        vals[m] = val; s += val;
        ce_f32[(size_t)row*128 + mm] = val;
    }
    const float mean = s * (1.0f/16.0f);
    float ss = 0.f;
    #pragma unroll
    for (int m = 0; m < 16; ++m) { float d = vals[m] - mean; ss += d*d; }
    const float stdv = sqrtf(ss * (1.0f/15.0f));
    const float inv  = 1.0f / (stdv + 1e-5f);
    #pragma unroll
    for (int m = 0; m < 16; ++m) {
        const int mm = hh*16 + m;
        ce_bf[(size_t)row*128 + mm] = (__bf16)(clw[mm]*(vals[m]-mean)*inv + clb[mm]);
    }
}

// ---------------------------------------------------------------------------
// Generic bf16 WMMA GEMM: out = ACT(A @ W^T + bias) + resScale*res
// B-tile (64x32) staged per K-step via GLOBAL_LOAD_ASYNC_TO_LDS_B128;
// A-tile (128x32) staged per K-step via TDM tensor_load_to_lds (one issue by
// wave 0, TENSORcnt) when available, else direct per-wave global loads.
// Both double-buffered: operand movement for step s+1 overlaps WMMA of step s.
// 8 waves/block, each wave a 16x64 strip (4 v_wmma per K-step).
// ---------------------------------------------------------------------------
template<int ACT, int RES, int OUTBF>
__global__ __launch_bounds__(256)
void gemm_wmma(const __bf16* __restrict__ A, const __bf16* __restrict__ W,
               const float* __restrict__ bias, const float* __restrict__ res,
               float resScale, float* __restrict__ outF, __bf16* __restrict__ outB,
               int M, int N, int K, int rowXor)
{
    __shared__ __bf16 bTile[2][64*32];       // [buf][col*32 + k] : 2 x 4KB
#ifdef HAVE_TDM
    __shared__ __bf16 aTile[2][128*32];      // [buf][row*32 + k] : 2 x 8KB
#endif
    const int t    = threadIdx.x;
    const int wave = t >> 5;
    const int lane = t & 31;
    const int l    = lane & 15;
    const int hl   = lane >> 4;
    const int m0   = blockIdx.x * 128 + wave * 16;
    const int n0   = blockIdx.y * 64;

    // staging role: each thread copies one 16B chunk of the 64x32 B tile
    const int cr = t >> 2;                   // column 0..63
    const int cc = t & 3;                    // 8-element K chunk 0..3
    const __bf16* wSrc = W + (size_t)(n0 + cr) * K + cc * 8;
    __bf16* wDst0 = &bTile[0][cr * 32 + cc * 8];
    __bf16* wDst1 = &bTile[1][cr * 32 + cc * 8];

#ifdef HAVE_TDM
    // block-uniform A tile base: rowXor (2^15) flips a bit above the 128-row
    // block, so base^rowXor + i stays a contiguous 128-row range.
    const __bf16* aSrc = A + (size_t)((blockIdx.x * 128) ^ rowXor) * K;
    if (wave == 0) tdm_load_tile_128x32(&aTile[0][0], aSrc, (unsigned)K);
#else
    const int aRow = (m0 + l) ^ rowXor;
    const __bf16* aBase = A + (size_t)aRow * K + hl * 8;
#endif
    async_ld_b128(wDst0, wSrc);              // prologue: tile 0 in flight

    v8f acc[4] = {};
    const int S = K >> 5;
    for (int s = 0; s < S; ++s) {
        const int k0 = s << 5;
#ifdef HAVE_TDM
        if (wave == 0) wait_tensor0();       // A tile s landed (wave 0 issued)
#endif
        wait_async0();                       // my chunk of B tile s landed
        __syncthreads();                     // all chunks landed; prev reads done
        if (s + 1 < S) {                     // overlap copies(s+1) with compute(s)
#ifdef HAVE_TDM
            if (wave == 0)
                tdm_load_tile_128x32(&aTile[(s + 1) & 1][0], aSrc + k0 + 32, (unsigned)K);
#endif
            async_ld_b128((s & 1) ? wDst0 : wDst1, wSrc + k0 + 32);
        }

        BF16x16 a;
#ifdef HAVE_TDM
        const __bf16* ar = &aTile[s & 1][(wave * 16 + l) * 32 + hl * 8];
        a.h[0] = *(const v8bf*)(ar);
        a.h[1] = *(const v8bf*)(ar + 16);
#else
        a.h[0] = *(const v8bf*)(aBase + k0);
        a.h[1] = *(const v8bf*)(aBase + k0 + 16);
#endif
        const __bf16* bb = &bTile[s & 1][l * 32 + hl * 8];
        #pragma unroll
        for (int j = 0; j < 4; ++j) {
            BF16x16 b;
            b.h[0] = *(const v8bf*)(bb + j * 16 * 32);
            b.h[1] = *(const v8bf*)(bb + j * 16 * 32 + 16);
            acc[j] = __builtin_amdgcn_wmma_f32_16x16x32_bf16(
                false, a.v, false, b.v, (short)0, acc[j], false, false);
        }
    }
    #pragma unroll
    for (int j = 0; j < 4; ++j) {
        const int col = n0 + j*16 + l;
        const float bv = bias ? bias[col] : 0.0f;
        #pragma unroll
        for (int r = 0; r < 8; ++r) {
            const int row = m0 + hl*8 + r;
            float vv = acc[j][r] + bv;
            if (ACT == 1) vv = gelu_exact(vv);
            if (RES == 1) vv += resScale * res[(size_t)row * N + col];
            if (OUTBF) outB[(size_t)row * N + col] = (__bf16)vv;
            else       outF[(size_t)row * N + col] = vv;
        }
    }
}

// ---------------------------------------------------------------------------
// Rank-32 reductions: ktv[bh] += k^T v, ktc[bh] += k^T ce over a split of N.
// Stage 32x(32+32+16) bf16 tiles into LDS via async b32 copies; waves 0-3 do
// the four 16x16 ktv tiles, waves 4-5 the two 16x16 ktc tiles (WMMA, K=32).
// ---------------------------------------------------------------------------
__global__ __launch_bounds__(256)
void ktv_kernel(const __bf16* __restrict__ k_bf, const __bf16* __restrict__ v_bf,
                const __bf16* __restrict__ ce_bf,
                float* __restrict__ ktv_acc, float* __restrict__ ktc_acc)
{
    __shared__ __bf16 k_lds[32*32];
    __shared__ __bf16 v_lds[32*32];
    __shared__ __bf16 ce_lds[32*16];
    const int bh = blockIdx.x;
    const int bb = bh >> 3, hh = bh & 7;
    const int t = threadIdx.x, wave = t >> 5, lane = t & 31;
    const int l = lane & 15, hl = lane >> 4;
    const size_t rowBase = (size_t)bb * 16384 + (size_t)blockIdx.y * 1024;
    const int mt = (wave < 4) ? (wave >> 1) : (wave - 4);
    const int nt = wave & 1;
    const int i0 = t, i1 = t + 256;
    v8f acc = {};
    for (int step = 0; step < 32; ++step) {
        const size_t n0 = rowBase + step * 32;
        __syncthreads();                     // previous tile fully consumed
        {
            const unsigned int* kp = (const unsigned int*)(k_bf + n0*256 + hh*32);
            const unsigned int* vp = (const unsigned int*)(v_bf + n0*256 + hh*32);
            const unsigned int* cp = (const unsigned int*)(ce_bf + n0*128 + hh*16);
            unsigned int* kl = (unsigned int*)k_lds;
            unsigned int* vl = (unsigned int*)v_lds;
            unsigned int* cl = (unsigned int*)ce_lds;
            async_ld_b32(&kl[i0], &kp[(i0 >> 4) * 128 + (i0 & 15)]);
            async_ld_b32(&kl[i1], &kp[(i1 >> 4) * 128 + (i1 & 15)]);
            async_ld_b32(&vl[i0], &vp[(i0 >> 4) * 128 + (i0 & 15)]);
            async_ld_b32(&vl[i1], &vp[(i1 >> 4) * 128 + (i1 & 15)]);
            async_ld_b32(&cl[i0], &cp[(i0 >> 3) * 64 + (i0 & 7)]);
        }
        wait_async0();
        __syncthreads();
        if (wave < 6) {
            BF16x16 a, b;
            #pragma unroll
            for (int e = 0; e < 16; ++e) {       // A = k^T (M=d, K=n)
                const int kk = hl * 16 + e;
                a.e[e] = k_lds[kk * 32 + mt * 16 + l];
            }
            if (wave < 4) {
                #pragma unroll
                for (int e = 0; e < 16; ++e)
                    b.e[e] = v_lds[(hl * 16 + e) * 32 + nt * 16 + l];
            } else {
                #pragma unroll
                for (int e = 0; e < 16; ++e)
                    b.e[e] = ce_lds[(hl * 16 + e) * 16 + l];
            }
            acc = __builtin_amdgcn_wmma_f32_16x16x32_bf16(
                false, a.v, false, b.v, (short)0, acc, false, false);
        }
    }
    if (wave < 4) {
        #pragma unroll
        for (int r = 0; r < 8; ++r) {
            const int d = mt*16 + hl*8 + r, e = nt*16 + l;
            atomicAdd(&ktv_acc[bh*1024 + d*32 + e], acc[r]);
        }
    } else if (wave < 6) {
        #pragma unroll
        for (int r = 0; r < 8; ++r) {
            const int d = mt*16 + hl*8 + r, e = l;
            atomicAdd(&ktc_acc[bh*512 + d*16 + e], acc[r]);
        }
    }
}

// Scale by 1/N and store transposed bf16 (B-fragment friendly layout).
__global__ __launch_bounds__(256)
void ktv_fin_kernel(const float* __restrict__ ktv_acc, const float* __restrict__ ktc_acc,
                    __bf16* __restrict__ ktvT, __bf16* __restrict__ ktcT)
{
    const int gid = blockIdx.x * blockDim.x + threadIdx.x;   // 49152
    const float sc = 1.0f / 16384.0f;
    if (gid < 32768) {
        const int bh = gid >> 10, rem = gid & 1023;
        const int d = rem >> 5, e = rem & 31;
        ktvT[bh*1024 + e*32 + d] = (__bf16)(ktv_acc[gid] * sc);
    } else {
        const int g2 = gid - 32768;
        const int bh = g2 >> 9, rem = g2 & 511;
        const int d = rem >> 4, e = rem & 15;
        ktcT[bh*512 + e*32 + d] = (__bf16)(ktc_acc[g2] * sc);
    }
}

// ---------------------------------------------------------------------------
// app = q @ ktv and crev = q @ ktc per head; fuse t = app + xn (bf16 out) and
// dm = crev - ce_flat (bf16 out). One wave per head, 16-row tile, K=32 exact.
// ---------------------------------------------------------------------------
__global__ __launch_bounds__(256)
void app_crev_kernel(const __bf16* __restrict__ qm, const __bf16* __restrict__ ktvT,
                     const __bf16* __restrict__ ktcT, const float* __restrict__ xn,
                     const float* __restrict__ ce_f32,
                     __bf16* __restrict__ t_bf, __bf16* __restrict__ dm_bf)
{
    const int hh   = threadIdx.x >> 5;
    const int lane = threadIdx.x & 31;
    const int l    = lane & 15, hl = lane >> 4;
    const int m0   = blockIdx.x * 16;
    const int bh   = (m0 >> 14) * 8 + hh;

    BF16x16 a;
    const __bf16* ap = qm + (size_t)(m0 + l) * 256 + hh * 32 + hl * 16;
    a.h[0] = *(const v8bf*)ap;
    a.h[1] = *(const v8bf*)(ap + 8);

    const __bf16* bvBase = ktvT + bh * 1024 + hl * 16;
    BF16x16 b0, b1, b2;
    b0.h[0] = *(const v8bf*)(bvBase + l * 32);
    b0.h[1] = *(const v8bf*)(bvBase + l * 32 + 8);
    b1.h[0] = *(const v8bf*)(bvBase + (16 + l) * 32);
    b1.h[1] = *(const v8bf*)(bvBase + (16 + l) * 32 + 8);
    const __bf16* bcBase = ktcT + bh * 512 + l * 32 + hl * 16;
    b2.h[0] = *(const v8bf*)(bcBase);
    b2.h[1] = *(const v8bf*)(bcBase + 8);

    v8f accA0 = {}, accA1 = {}, accC = {};
    accA0 = __builtin_amdgcn_wmma_f32_16x16x32_bf16(false, a.v, false, b0.v, (short)0, accA0, false, false);
    accA1 = __builtin_amdgcn_wmma_f32_16x16x32_bf16(false, a.v, false, b1.v, (short)0, accA1, false, false);
    accC  = __builtin_amdgcn_wmma_f32_16x16x32_bf16(false, a.v, false, b2.v, (short)0, accC,  false, false);

    #pragma unroll
    for (int r = 0; r < 8; ++r) {
        const int row = m0 + hl * 8 + r;
        {
            const int col = hh * 32 + l;
            t_bf[(size_t)row*256 + col] = (__bf16)(accA0[r] + xn[(size_t)row*256 + col]);
        }
        {
            const int col = hh * 32 + 16 + l;
            t_bf[(size_t)row*256 + col] = (__bf16)(accA1[r] + xn[(size_t)row*256 + col]);
        }
        {
            const int colM = hh * 16 + l;
            dm_bf[(size_t)row*128 + colM] = (__bf16)(accC[r] - ce_f32[(size_t)row*128 + colM]);
        }
    }
}

// ---------------------------------------------------------------------------
// Depthwise 3x3 SAME conv (channel-last) + bias + exact GELU.
// ---------------------------------------------------------------------------
__global__ __launch_bounds__(256)
void dwconv_kernel(const __bf16* __restrict__ h1, const float* __restrict__ wgt,
                   const float* __restrict__ bias, __bf16* __restrict__ h2)
{
    const size_t gid = (size_t)blockIdx.x * blockDim.x + threadIdx.x;  // 4*16384*768
    const int c = (int)(gid % 768);
    const size_t pix = gid / 768;
    const int xx = (int)(pix % 128);
    const int yy = (int)((pix / 128) % 128);
    const int bb = (int)(pix / 16384);
    float acc = bias[c];
    #pragma unroll
    for (int dy = -1; dy <= 1; ++dy) {
        const int y2 = yy + dy;
        if (y2 < 0 || y2 > 127) continue;
        #pragma unroll
        for (int dx = -1; dx <= 1; ++dx) {
            const int x2 = xx + dx;
            if (x2 < 0 || x2 > 127) continue;
            const size_t idx = ((size_t)bb*16384 + (size_t)y2*128 + x2) * 768 + c;
            acc += (float)h1[idx] * wgt[c*9 + (dy+1)*3 + (dx+1)];
        }
    }
    h2[gid] = (__bf16)gelu_exact(acc);
}

__global__ __launch_bounds__(256)
void f2bf_kernel(const float* __restrict__ s, __bf16* __restrict__ d, int n)
{
    const int i = blockIdx.x * blockDim.x + threadIdx.x;
    if (i < n) d[i] = (__bf16)s[i];
}

// ---------------------------------------------------------------------------
extern "C" void kernel_launch(void* const* d_in, const int* in_sizes, int n_in,
                              void* d_out, int out_size, void* d_ws, size_t ws_size,
                              hipStream_t stream)
{
    (void)in_sizes; (void)n_in; (void)out_size; (void)ws_size;
    const float* x       = (const float*)d_in[0];
    const float* cor     = (const float*)d_in[1];
    const float* q_w     = (const float*)d_in[2];
    const float* kv_w    = (const float*)d_in[3];
    const float* cor_w   = (const float*)d_in[4];
    const float* kln_w   = (const float*)d_in[5];
    const float* kln_b   = (const float*)d_in[6];
    const float* vln_w   = (const float*)d_in[7];
    const float* vln_b   = (const float*)d_in[8];
    const float* corln_w = (const float*)d_in[9];
    const float* corln_b = (const float*)d_in[10];
    const float* op1_w   = (const float*)d_in[11];
    const float* op1_b   = (const float*)d_in[12];
    const float* op2_w   = (const float*)d_in[13];
    const float* op2_b   = (const float*)d_in[14];
    const float* om1_w   = (const float*)d_in[15];
    const float* om1_b   = (const float*)d_in[16];
    const float* om2_w   = (const float*)d_in[17];
    const float* om2_b   = (const float*)d_in[18];
    const float* norm1_w = (const float*)d_in[19];
    const float* norm1_b = (const float*)d_in[20];
    const float* norm2_w = (const float*)d_in[21];
    const float* norm2_b = (const float*)d_in[22];
    const float* fc1_w   = (const float*)d_in[23];
    const float* fc1_b   = (const float*)d_in[24];
    const float* dw_w    = (const float*)d_in[25];
    const float* dw_b    = (const float*)d_in[26];
    const float* fc2_w   = (const float*)d_in[27];
    const float* fc2_b   = (const float*)d_in[28];

    float* out    = (float*)d_out;
    float* x_out  = out;                 // 65536*256 (also staged as x_back)
    float* motion = out + 16777216;      // 65536*128

    char* ws = (char*)d_ws;
    const size_t MB = 1ull << 20;
    // lifetime-overlaid workspace (~435 MB)
    float*  xn_f32 = (float*) (ws + 0*MB);     // alive through op2 residual
    __bf16* xn_bf  = (__bf16*)(ws + 64*MB);    // dead after kv GEMM
    __bf16* dm_bf  = (__bf16*)(ws + 64*MB);    //   overlays xn_bf
    __bf16* g_bf   = (__bf16*)(ws + 80*MB);
    __bf16* q_bf   = (__bf16*)(ws + 96*MB);    // dead after app_crev
    __bf16* y_bf   = (__bf16*)(ws + 96*MB);    //   overlays q_bf
    float*  kv_f32 = (float*) (ws + 128*MB);   // dead after lnhead
    __bf16* h1_bf  = (__bf16*)(ws + 128*MB);   //   overlays kv_f32
    __bf16* k_bf   = (__bf16*)(ws + 256*MB);   // dead after ktv
    __bf16* v_bf   = (__bf16*)(ws + 288*MB);
    float*  ce_f32 = (float*) (ws + 320*MB);   // dead after app_crev
    __bf16* h2_bf  = (__bf16*)(ws + 256*MB);   //   overlays k,v,ce (96 MB)
    __bf16* ce_bf  = (__bf16*)(ws + 352*MB);   // dead after ktv
    __bf16* t_bf   = (__bf16*)(ws + 368*MB);
    __bf16* u_bf   = (__bf16*)(ws + 400*MB);
    float*  ktv_acc = (float*)(ws + 432*MB);
    float*  ktc_acc = ktv_acc + 32*1024;
    __bf16* ktvT    = (__bf16*)(ktc_acc + 32*512);
    __bf16* ktcT    = ktvT + 32*1024;
    __bf16* wpool   = (__bf16*)(ws + 433*MB);
    __bf16* qw_bf   = wpool;
    __bf16* kvw_bf  = qw_bf  + 65536;
    __bf16* op1w_bf = kvw_bf + 131072;
    __bf16* op2w_bf = op1w_bf + 65536;
    __bf16* om1w_bf = op2w_bf + 65536;
    __bf16* om2w_bf = om1w_bf + 16384;
    __bf16* fc1w_bf = om2w_bf + 16384;
    __bf16* fc2w_bf = fc1w_bf + 196608;

    // weight conversion (tiny, fully deterministic each call)
    f2bf_kernel<<<256, 256, 0, stream>>>(q_w,   qw_bf,   65536);
    f2bf_kernel<<<512, 256, 0, stream>>>(kv_w,  kvw_bf,  131072);
    f2bf_kernel<<<256, 256, 0, stream>>>(op1_w, op1w_bf, 65536);
    f2bf_kernel<<<256, 256, 0, stream>>>(op2_w, op2w_bf, 65536);
    f2bf_kernel<<<64,  256, 0, stream>>>(om1_w, om1w_bf, 16384);
    f2bf_kernel<<<64,  256, 0, stream>>>(om2_w, om2w_bf, 16384);
    f2bf_kernel<<<768, 256, 0, stream>>>(fc1_w, fc1w_bf, 196608);
    f2bf_kernel<<<768, 256, 0, stream>>>(fc2_w, fc2w_bf, 196608);

    // 1. xn = LN1(x) -> f32 + bf16
    ln256_kernel<<<8192, 256, 0, stream>>>(x, norm1_w, norm1_b, xn_f32, xn_bf);
    // 2. ce_flat + per-head LN(ce)
    ce_kernel<<<2048, 256, 0, stream>>>(cor, cor_w, corln_w, corln_b, ce_f32, ce_bf);
    // 3. q = xn @ q_w^T
    gemm_wmma<0,0,1><<<dim3(512, 4), 256, 0, stream>>>(
        xn_bf, qw_bf, nullptr, nullptr, 0.f, nullptr, q_bf, 65536, 256, 256, 0);
    // 4. kv = xr @ kv_w^T  (xr = frame-half swap via rowXor)
    gemm_wmma<0,0,0><<<dim3(512, 8), 256, 0, stream>>>(
        xn_bf, kvw_bf, nullptr, nullptr, 0.f, kv_f32, nullptr, 65536, 512, 256, 32768);
    // 5. k,v = custom_ln per head
    lnhead_kernel<<<4096, 256, 0, stream>>>(kv_f32, kln_w, kln_b, vln_w, vln_b, k_bf, v_bf);
    // 6. ktv = k^T v / N, ktc = k^T ce / N  (WMMA split-N + f32 atomics)
    (void)hipMemsetAsync(ktv_acc, 0, (32*1024 + 32*512) * sizeof(float), stream);
    ktv_kernel<<<dim3(32, 16), 256, 0, stream>>>(k_bf, v_bf, ce_bf, ktv_acc, ktc_acc);
    ktv_fin_kernel<<<192, 256, 0, stream>>>(ktv_acc, ktc_acc, ktvT, ktcT);
    // 7. t = q@ktv + xn, dm = q@ktc - ce_flat
    app_crev_kernel<<<4096, 256, 0, stream>>>(q_bf, ktvT, ktcT, xn_f32, ce_f32, t_bf, dm_bf);
    // 8. u = gelu(t @ op1^T + b1)
    gemm_wmma<1,0,1><<<dim3(512, 4), 256, 0, stream>>>(
        t_bf, op1w_bf, op1_b, nullptr, 0.f, nullptr, u_bf, 65536, 256, 256, 0);
    // 9. x_back = u @ op2^T + b2 + 2*xn  (staged in d_out's x_out region)
    gemm_wmma<0,1,0><<<dim3(512, 4), 256, 0, stream>>>(
        u_bf, op2w_bf, op2_b, xn_f32, 2.0f, x_out, nullptr, 65536, 256, 256, 0);
    // 10. motion branch
    gemm_wmma<1,0,1><<<dim3(512, 2), 256, 0, stream>>>(
        dm_bf, om1w_bf, om1_b, nullptr, 0.f, nullptr, g_bf, 65536, 128, 128, 0);
    gemm_wmma<0,0,0><<<dim3(512, 2), 256, 0, stream>>>(
        g_bf, om2w_bf, om2_b, nullptr, 0.f, motion, nullptr, 65536, 128, 128, 0);
    // 11. y = LN2(x_back)
    ln256_kernel<<<8192, 256, 0, stream>>>(x_out, norm2_w, norm2_b, nullptr, y_bf);
    // 12. h1 = y @ fc1^T + b
    gemm_wmma<0,0,1><<<dim3(512, 12), 256, 0, stream>>>(
        y_bf, fc1w_bf, fc1_b, nullptr, 0.f, nullptr, h1_bf, 65536, 768, 256, 0);
    // 13. depthwise 3x3 + bias + gelu
    dwconv_kernel<<<196608, 256, 0, stream>>>(h1_bf, dw_w, dw_b, h2_bf);
    // 14. x_out = h2 @ fc2^T + b + x
    gemm_wmma<0,1,0><<<dim3(512, 4), 256, 0, stream>>>(
        h2_bf, fc2w_bf, fc2_b, x, 1.0f, x_out, nullptr, 65536, 256, 768, 0);
}
